// EnsembleDynamicsModel_15006615733146
// MI455X (gfx1250) — matrix operand order repaired and checked by reading, not compiled, not run on hardware
//
#include <hip/hip_runtime.h>

// ---------------------------------------------------------------------------
// Ensemble dynamics model forward (PETS-style), bf16 WMMA on gfx1250.
//   E=7, B=32768, D_IN=40 (pad 64), HIDDEN=200 (pad N=208, K=224), OUT=66 (pad 80)
//   Each wave handles 32 batch rows (2 M-tiles) so every B fragment feeds 2 WMMAs.
//   t-loop kept rolled so A fragments stay register-resident (no LDS re-reads).
// ---------------------------------------------------------------------------

#define EN      7
#define BATCH   32768
#define HID     200
#define OUTD    66
#define SDIM    32
#define ADIM    8

#define LSTRIDE 232          // bf16 elems per LDS row (mult of 8 -> 16B aligned rows,
                             // 116 dwords -> 16 distinct banks, b128 ranges disjoint)
#define NT_H    13           // hidden N tiles (208)
#define KS_H    7            // hidden K steps  (224)
#define KS_IN   2            // input  K steps  (64)
#define NT_O    5            // output N tiles  (80)
#define MT      2            // M tiles per wave (32 rows)
#define ROWS    (MT * 16)

typedef __attribute__((ext_vector_type(16))) __bf16 v16bf;
typedef __attribute__((ext_vector_type(8)))  float  v8f;

union Frag { v16bf v; uint4 q[2]; };

__device__ __forceinline__ unsigned short f2bf(float f) {
    union { float f; unsigned int u; } x; x.f = f;
    unsigned int u = x.u;
    u += 0x7FFFu + ((u >> 16) & 1u);          // round-to-nearest-even
    return (unsigned short)(u >> 16);
}

// Fast sigmoid/swish using hardware v_rcp_f32 (avoids IEEE div expansion).
__device__ __forceinline__ float swish_f(float x) {
    return x * __builtin_amdgcn_rcpf(1.0f + __expf(-x));
}

__device__ __forceinline__ float softplus_f(float x) {
    return (x > 20.0f) ? x : __logf(1.0f + __expf(x));
}

// ---------------------------------------------------------------------------
// Weight pre-pack: fp32 (E,K,N) -> bf16 B-fragment order
//   packed[e][ntile][kstep][lane][j] ;  lane<16: N=t*16+lane,   K=s*32+j (j<8), s*32+8+j (j>=8)
//                                       lane>=16: N=t*16+lane-16, K=s*32+8+j (j<8), s*32+16+j (j>=8)
// Zero-padded outside (Kreal, Nreal).
// ---------------------------------------------------------------------------
__global__ void pack_w_kernel(const float* __restrict__ W, unsigned short* __restrict__ dst,
                              int Kreal, int Nreal, int kSteps, int nTiles) {
    int idx = blockIdx.x * blockDim.x + threadIdx.x;
    int per_e = nTiles * kSteps * 512;
    if (idx >= EN * per_e) return;
    int e    = idx / per_e;
    int rem  = idx % per_e;
    int t    = rem / (kSteps * 512);
    int rem2 = rem % (kSteps * 512);
    int s    = rem2 / 512;
    int rem3 = rem2 % 512;
    int L    = rem3 >> 4;
    int j    = rem3 & 15;
    int n    = t * 16 + (L & 15);
    int k    = s * 32 + ((L < 16) ? 0 : 8) + ((j < 8) ? j : j + 8);
    float v  = (k < Kreal && n < Nreal) ? W[((size_t)e * Kreal + k) * Nreal + n] : 0.0f;
    dst[idx] = f2bf(v);
}

// ---------------------------------------------------------------------------
// One linear layer for a wave's 32-row tile held in its LDS slice.
//   A fragments (2 x 16x32 bf16) from own LDS rows, held in VGPRs all layer;
//   B fragments streamed from packed weights (L2-resident), reused by MT WMMAs.
//   Accumulators start at inline 0; bias is folded into the activation phase.
//   LAST=false: swish + bf16 -> LDS.   LAST=true: f32 -> LDS (stride 84).
// ---------------------------------------------------------------------------
template <int NT, int KS, bool LAST>
__device__ __forceinline__ void layer_fwd(const unsigned short* __restrict__ Wp,
                                          const float* __restrict__ bias, int Nreal,
                                          unsigned short* slice, int lane) {
    Frag a[MT][KS];
    {
        int m  = lane & 15;
        int k0 = (lane < 16) ? 0 : 8;
#pragma unroll
        for (int mt = 0; mt < MT; ++mt) {
            const unsigned short* arow = slice + (mt * 16 + m) * LSTRIDE + k0;
#pragma unroll
            for (int s = 0; s < KS; ++s) {
                const unsigned short* p = arow + s * 32;
                a[mt][s].q[0] = *(const uint4*)(p);        // K = k0 .. k0+7
                a[mt][s].q[1] = *(const uint4*)(p + 16);   // K = k0+16 .. k0+23
            }
        }
    }
    int nl   = lane & 15;
    int mrow = (lane < 16) ? 0 : 8;
#pragma unroll 1
    for (int t = 0; t < NT; ++t) {
        int n    = t * 16 + nl;
        float bv = (n < Nreal) ? bias[n] : 0.0f;
        v8f c[MT];
#pragma unroll
        for (int mt = 0; mt < MT; ++mt)
            c[mt] = (v8f){ 0.f, 0.f, 0.f, 0.f, 0.f, 0.f, 0.f, 0.f };
#pragma unroll
        for (int s = 0; s < KS; ++s) {
            Frag b;
            const unsigned short* wp = Wp + ((size_t)((t * KS + s) * 32 + lane) << 4);
            b.q[0] = *(const uint4*)(wp);
            b.q[1] = *(const uint4*)(wp + 8);
#pragma unroll
            for (int mt = 0; mt < MT; ++mt)
                c[mt] = __builtin_amdgcn_wmma_f32_16x16x32_bf16(
                            false, a[mt][s].v, false, b.v, (short)0, c[mt], false, false);
        }
        if (!LAST) {
#pragma unroll
            for (int mt = 0; mt < MT; ++mt) {
                unsigned short* dst = slice + mt * 16 * LSTRIDE + n;
#pragma unroll
                for (int v = 0; v < 8; ++v)
                    dst[(v + mrow) * LSTRIDE] = f2bf(swish_f(c[mt][v] + bv));
            }
        } else {
            float* of = (float*)slice;
#pragma unroll
            for (int mt = 0; mt < MT; ++mt)
#pragma unroll
                for (int v = 0; v < 8; ++v)
                    of[(mt * 16 + v + mrow) * 84 + n] = c[mt][v] + bv;
        }
    }
}

// ---------------------------------------------------------------------------
// Main forward kernel: grid = EN * (BATCH/128), 128 threads (4 waves x 32 rows)
// ---------------------------------------------------------------------------
__global__ __launch_bounds__(128) void ensemble_fwd_kernel(
    const float* __restrict__ state, const float* __restrict__ action,
    const float* __restrict__ noise, const unsigned short* __restrict__ wsb,
    size_t o1, size_t o2, size_t o3, size_t o4, size_t o5,
    const float* __restrict__ b1, const float* __restrict__ b2,
    const float* __restrict__ b3, const float* __restrict__ b4,
    const float* __restrict__ b5, const float* __restrict__ max_lv,
    const float* __restrict__ min_lv, float* __restrict__ out) {

    const int lane = threadIdx.x & 31;
    const int wave = threadIdx.x >> 5;
    const int blocksPerE = BATCH / (4 * ROWS);
    const int e  = blockIdx.x / blocksPerE;
    const int r0 = (blockIdx.x % blocksPerE) * (4 * ROWS) + wave * ROWS;

    __shared__ unsigned short lds[4 * ROWS * LSTRIDE];
    unsigned short* slice = lds + wave * ROWS * LSTRIDE;

    // Zero the slice (padded K columns must be exactly 0 for all layers).
    {
        unsigned int* sz = (unsigned int*)slice;
        for (int i = lane; i < ROWS * LSTRIDE / 2; i += 32) sz[i] = 0;
    }
    // Stage x = concat(state, action) as bf16 rows (cols 0..39).
    for (int i = lane; i < ROWS * 40; i += 32) {
        int r = i / 40, cidx = i % 40;
        float v = (cidx < SDIM) ? state[(size_t)(r0 + r) * SDIM + cidx]
                                : action[(size_t)(r0 + r) * ADIM + (cidx - SDIM)];
        slice[r * LSTRIDE + cidx] = f2bf(v);
    }

    const size_t W1_per = (size_t)NT_H * KS_IN * 512;
    const size_t WH_per = (size_t)NT_H * KS_H * 512;
    const size_t W5_per = (size_t)NT_O * KS_H * 512;

    // Layer 1: 40(->64) -> 200
    layer_fwd<NT_H, KS_IN, false>(wsb + o1 + (size_t)e * W1_per, b1 + e * HID, HID, slice, lane);

    // Layers 2..4: 200(->224) -> 200
    const unsigned short* WH[3] = { wsb + o2 + (size_t)e * WH_per,
                                    wsb + o3 + (size_t)e * WH_per,
                                    wsb + o4 + (size_t)e * WH_per };
    const float* bH[3] = { b2 + e * HID, b3 + e * HID, b4 + e * HID };
#pragma unroll 1
    for (int l = 0; l < 3; ++l)
        layer_fwd<NT_H, KS_H, false>(WH[l], bH[l], HID, slice, lane);

    // Layer 5: 200(->224) -> 66(->80), f32 to LDS
    layer_fwd<NT_O, KS_H, true>(wsb + o5 + (size_t)e * W5_per, b5 + e * OUTD, OUTD, slice, lane);

    // Epilogue: logvar clamp via softplus, sample, write next_states + rewards.
    const float* of = (const float*)slice;
    for (int i = lane; i < ROWS * 33; i += 32) {
        int r = i / 33, cidx = i % 33;
        int brow = r0 + r;
        float m  = of[r * 84 + cidx];
        float lv = of[r * 84 + 33 + cidx];
        float mx = max_lv[cidx], mn = min_lv[cidx];
        lv = mx - softplus_f(mx - lv);
        lv = mn + softplus_f(lv - mn);
        float sd  = __expf(0.5f * lv);
        float smp = m + sd * noise[((size_t)e * BATCH + brow) * 33 + cidx];
        if (cidx < SDIM)
            out[((size_t)e * BATCH + brow) * SDIM + cidx] =
                state[(size_t)brow * SDIM + cidx] + smp;
        else
            out[(size_t)EN * BATCH * SDIM + (size_t)e * BATCH + brow] = smp;
    }
}

// ---------------------------------------------------------------------------
extern "C" void kernel_launch(void* const* d_in, const int* in_sizes, int n_in,
                              void* d_out, int out_size, void* d_ws, size_t ws_size,
                              hipStream_t stream) {
    (void)in_sizes; (void)n_in; (void)out_size; (void)ws_size;

    const float* state  = (const float*)d_in[0];
    const float* action = (const float*)d_in[1];
    const float* noise  = (const float*)d_in[2];
    const float* W1 = (const float*)d_in[3];  const float* b1 = (const float*)d_in[4];
    const float* W2 = (const float*)d_in[5];  const float* b2 = (const float*)d_in[6];
    const float* W3 = (const float*)d_in[7];  const float* b3 = (const float*)d_in[8];
    const float* W4 = (const float*)d_in[9];  const float* b4 = (const float*)d_in[10];
    const float* W5 = (const float*)d_in[11]; const float* b5 = (const float*)d_in[12];
    const float* max_lv = (const float*)d_in[13];
    const float* min_lv = (const float*)d_in[14];
    float* out = (float*)d_out;

    unsigned short* ws = (unsigned short*)d_ws;
    const size_t W1_per = (size_t)NT_H * KS_IN * 512;   // 13312 bf16 / member
    const size_t WH_per = (size_t)NT_H * KS_H  * 512;   // 46592
    const size_t W5_per = (size_t)NT_O * KS_H  * 512;   // 17920
    size_t o1 = 0;
    size_t o2 = o1 + EN * W1_per;
    size_t o3 = o2 + EN * WH_per;
    size_t o4 = o3 + EN * WH_per;
    size_t o5 = o4 + EN * WH_per;                       // total ~2.4 MB of d_ws

    auto pack = [&](const float* W, size_t dstOff, int Kreal, int Nreal, int kS, int nT) {
        int total = EN * nT * kS * 512;
        pack_w_kernel<<<(total + 255) / 256, 256, 0, stream>>>(W, ws + dstOff,
                                                               Kreal, Nreal, kS, nT);
    };
    pack(W1, o1, 40,  HID,  KS_IN, NT_H);
    pack(W2, o2, HID, HID,  KS_H,  NT_H);
    pack(W3, o3, HID, HID,  KS_H,  NT_H);
    pack(W4, o4, HID, HID,  KS_H,  NT_O == 0 ? 0 : NT_H);
    pack(W5, o5, HID, OUTD, KS_H,  NT_O);

    ensemble_fwd_kernel<<<EN * (BATCH / (4 * ROWS)), 128, 0, stream>>>(
        state, action, noise, ws, o1, o2, o3, o4, o5,
        b1, b2, b3, b4, b5, max_lv, min_lv, out);
}
